// SmolVLM2VisionAttention_60773787238428
// MI455X (gfx1250) — compile-verified, matrix operand order checked by reading, not run on hardware
//
#include <hip/hip_runtime.h>
#include <stdint.h>

// ---------------- problem constants ----------------
#define Bn     16
#define Sq     729
#define SP     736            // S padded to multiple of 32 (23*32)
#define HID    1152
#define HEADS  16
#define DH     72
#define DP     96             // head dim padded to multiple of 32
#define DVP    80             // V^T row count padded to multiple of 16
#define MT     (Bn * Sq)      // 11664 total tokens
#define SCALE_C 0.1178511301977579f   // 72^-0.5

typedef __bf16 bf16_t;
typedef __attribute__((ext_vector_type(16))) __bf16 v16bf;
typedef __attribute__((ext_vector_type(8)))  float  v8f;

// round-to-nearest-even f32 -> bf16 (stored as raw u16)
__device__ __forceinline__ unsigned short f2bf(float x) {
    unsigned int u = __builtin_bit_cast(unsigned int, x);
    u += 0x7FFFu + ((u >> 16) & 1u);
    return (unsigned short)(u >> 16);
}

// Load a 16x32 bf16 A-fragment (or 32x16 B-fragment from a transposed,
// row-major matrix) following the CDNA5 16-bit WMMA VGPR layout:
//   lanes 0-15:  K = kbase+0..7   in VGPR0..3, K = kbase+16..23 in VGPR4..7
//   lanes 16-31: K = kbase+8..15  in VGPR0..3, K = kbase+24..31 in VGPR4..7
// -> two contiguous 16-byte loads per lane.
__device__ __forceinline__ v16bf load_frag(const unsigned short* base, int ld,
                                           int row, int kbase, int lane) {
    int koff = kbase + ((lane & 16) ? 8 : 0);
    const unsigned short* p = base + (size_t)row * ld + koff;
    union { uint4 u[2]; v16bf v; } t;
    t.u[0] = *reinterpret_cast<const uint4*>(p);        // K koff..koff+7
    t.u[1] = *reinterpret_cast<const uint4*>(p + 16);   // K koff+16..koff+23
    return t.v;
}

__device__ __forceinline__ v8f wmma_bf16(v16bf a, v16bf b, v8f c) {
    return __builtin_amdgcn_wmma_f32_16x16x32_bf16(false, a, false, b,
                                                   (short)0, c, false, false);
}

// ---------------- stage 0: pack / transpose / zero ----------------
__global__ __launch_bounds__(256) void pack_bf16(const float* __restrict__ in,
                                                 unsigned short* __restrict__ out,
                                                 int n) {
    int i = blockIdx.x * 256 + threadIdx.x;
    if (i < n) out[i] = f2bf(in[i]);
}

// Wt[n*HID + k] = bf16(W[k*HID + n])
__global__ __launch_bounds__(256) void transpose_pack(const float* __restrict__ W,
                                                      unsigned short* __restrict__ Wt) {
    int i = blockIdx.x * 256 + threadIdx.x;
    if (i < HID * HID) {
        int nrow = i / HID, k = i % HID;
        Wt[i] = f2bf(W[k * HID + nrow]);
    }
}

__global__ __launch_bounds__(256) void zero_u4(uint4* __restrict__ p, size_t n4) {
    size_t stride = (size_t)gridDim.x * 256;
    for (size_t i = blockIdx.x * 256 + threadIdx.x; i < n4; i += stride)
        p[i] = make_uint4(0u, 0u, 0u, 0u);
}

// ---------------- dense GEMM: C = A(bf16,[MT x HID]) @ Bt^T + bias -------
// mode 0: out bf16 -> [B][H][SP][DP]  (Q or K, head-split, padded)
// mode 1: out bf16 -> [B][H][DVP][SP] (V transposed per head, padded)
// mode 2: out f32  -> [MT x HID]      (final projection)
// block = 256 threads (8 waves, 4x2 wave grid), block tile 128(M) x 64(N),
// wave tile 32x32 (4 accumulators), K loop 36 x 32.
__global__ __launch_bounds__(256, 2) void gemm_bf16(
    const unsigned short* __restrict__ A,
    const unsigned short* __restrict__ Bt,
    const float* __restrict__ bias,
    void* __restrict__ Out, int mode) {
    int lane = threadIdx.x & 31;
    int w    = threadIdx.x >> 5;
    int m0 = blockIdx.x * 128 + (w & 3) * 32;
    int n0 = blockIdx.y * 64  + (w >> 2) * 32;

    const v8f zacc = {0.f,0.f,0.f,0.f,0.f,0.f,0.f,0.f};
    v8f acc[2][2];
    acc[0][0] = zacc; acc[0][1] = zacc; acc[1][0] = zacc; acc[1][1] = zacc;

    int r15 = lane & 15;
    int ma = m0 + r15;        if (ma > MT - 1) ma = MT - 1;   // clamp pad rows
    int mb = m0 + 16 + r15;   if (mb > MT - 1) mb = MT - 1;

    for (int kk = 0; kk < HID; kk += 32) {
        v16bf a0 = load_frag(A,  HID, ma,            kk, lane);
        v16bf a1 = load_frag(A,  HID, mb,            kk, lane);
        v16bf b0 = load_frag(Bt, HID, n0 + r15,      kk, lane);
        v16bf b1 = load_frag(Bt, HID, n0 + 16 + r15, kk, lane);
        acc[0][0] = wmma_bf16(a0, b0, acc[0][0]);
        acc[0][1] = wmma_bf16(a0, b1, acc[0][1]);
        acc[1][0] = wmma_bf16(a1, b0, acc[1][0]);
        acc[1][1] = wmma_bf16(a1, b1, acc[1][1]);
    }

    // C layout: lane n = lane&15 ; VGPR r -> m = r + (lane>=16 ? 8 : 0)
    int half = (lane >> 4) * 8;
    #pragma unroll
    for (int i = 0; i < 2; ++i) {
        #pragma unroll
        for (int j = 0; j < 2; ++j) {
            int nn = n0 + j * 16 + r15;
            float bv = bias[nn];
            #pragma unroll
            for (int r = 0; r < 8; ++r) {
                int mm = m0 + i * 16 + half + r;
                if (mm >= MT) continue;
                float val = acc[i][j][r] + bv;
                if (mode == 2) {
                    ((float*)Out)[(size_t)mm * HID + nn] = val;
                } else {
                    int b = mm / Sq, s = mm % Sq;
                    int h = nn / DH, d = nn % DH;
                    unsigned short* o = (unsigned short*)Out;
                    if (mode == 0)
                        o[((size_t)(b * HEADS + h) * SP + s) * DP + d] = f2bf(val);
                    else
                        o[((size_t)(b * HEADS + h) * DVP + d) * SP + s] = f2bf(val);
                }
            }
        }
    }
}

// ---------------- attention core: scores -> softmax -> PV ----------------
// grid = (23 query blocks, 256 b*h), block = 128 threads (4 waves)
// LDS: sc[32][736] f32 (94208 B) + pb[32][736] bf16 (47104 B) = 141312 B
__global__ __launch_bounds__(128, 2) void attn_kernel(
    const unsigned short* __restrict__ Qh,
    const unsigned short* __restrict__ Kh,
    const unsigned short* __restrict__ Vt,
    unsigned short* __restrict__ Ctx) {
    extern __shared__ char smem[];
    float*          sc = (float*)smem;                        // [32][SP]
    unsigned short* pb = (unsigned short*)(smem + 32 * SP * 4); // [32][SP]

    int lane = threadIdx.x & 31;
    int w    = threadIdx.x >> 5;
    int bh   = blockIdx.y;
    int q0   = blockIdx.x * 32;
    int r15  = lane & 15;
    const unsigned short* Qb = Qh + (size_t)bh * SP * DP;
    const unsigned short* Kb = Kh + (size_t)bh * SP * DP;
    const unsigned short* Vb = Vt + (size_t)bh * DVP * SP;

    // ---- phase 1: scores = Q K^T * scale, tiled 16x16, K = 96 (3 wmma) ----
    {
        int qt = w & 1;
        for (int kt = (w >> 1); kt < SP / 16; kt += 2) {
            v8f acc = {0.f,0.f,0.f,0.f,0.f,0.f,0.f,0.f};
            #pragma unroll
            for (int kk = 0; kk < DP; kk += 32) {
                v16bf a = load_frag(Qb, DP, q0 + qt * 16 + r15, kk, lane);
                v16bf b = load_frag(Kb, DP, kt * 16 + r15,      kk, lane);
                acc = wmma_bf16(a, b, acc);
            }
            int n  = kt * 16 + r15;
            int mb = qt * 16 + (lane >> 4) * 8;
            #pragma unroll
            for (int r = 0; r < 8; ++r)
                sc[(size_t)(mb + r) * SP + n] = acc[r] * SCALE_C;
        }
    }
    __syncthreads();

    // ---- phase 2: row softmax in f32, re-quantize probs to bf16 ----
    // wave w owns rows 8w..8w+7; 4 lanes per row, shuffle-reduce across them.
    {
        int rl = lane >> 2, j = lane & 3;
        float* row = sc + (size_t)(w * 8 + rl) * SP;
        float mx = -1e30f;
        for (int c = j; c < Sq; c += 4) mx = fmaxf(mx, row[c]);
        mx = fmaxf(mx, __shfl_xor(mx, 1, 32));
        mx = fmaxf(mx, __shfl_xor(mx, 2, 32));
        float sum = 0.f;
        for (int c = j; c < Sq; c += 4) sum += __expf(row[c] - mx);
        sum += __shfl_xor(sum, 1, 32);
        sum += __shfl_xor(sum, 2, 32);
        float inv = 1.0f / sum;
        unsigned short* prow = pb + (size_t)(w * 8 + rl) * SP;
        for (int c = j; c < SP; c += 4)
            prow[c] = (c < Sq) ? f2bf(__expf(row[c] - mx) * inv)
                               : (unsigned short)0;
    }
    __syncthreads();

    // ---- phase 3: out = P @ V  (A from LDS probs, B from V^T) ----
    {
        int b = bh >> 4, h = bh & 15;
        for (int t = w; t < 10; t += 4) {          // 2 q-tiles x 5 d-tiles
            int qt = t / 5, nt = t % 5;
            v8f acc = {0.f,0.f,0.f,0.f,0.f,0.f,0.f,0.f};
            for (int kk = 0; kk < SP; kk += 32) {
                v16bf a  = load_frag(pb, SP, qt * 16 + r15, kk, lane); // LDS
                v16bf bb = load_frag(Vb, SP, nt * 16 + r15, kk, lane);
                acc = wmma_bf16(a, bb, acc);
            }
            int d = nt * 16 + r15;
            if (d < DH) {
                #pragma unroll
                for (int r = 0; r < 8; ++r) {
                    int q = q0 + qt * 16 + (lane >> 4) * 8 + r;
                    if (q < Sq)
                        Ctx[((size_t)(b * Sq + q)) * HID + h * DH + d] =
                            f2bf(acc[r]);
                }
            }
        }
    }
}

// ---------------- host-side launch ----------------
extern "C" void kernel_launch(void* const* d_in, const int* in_sizes, int n_in,
                              void* d_out, int out_size, void* d_ws, size_t ws_size,
                              hipStream_t stream) {
    const float* X  = (const float*)d_in[0];
    const float* Wq = (const float*)d_in[1]; const float* bq = (const float*)d_in[2];
    const float* Wk = (const float*)d_in[3]; const float* bk = (const float*)d_in[4];
    const float* Wv = (const float*)d_in[5]; const float* bv = (const float*)d_in[6];
    const float* Wo = (const float*)d_in[7]; const float* bo = (const float*)d_in[8];

    // workspace layout (bytes, all 16B aligned)
    const size_t szX  = (size_t)MT * HID * 2;                 // 26,873,856
    const size_t szW  = (size_t)HID * HID * 2;                //  2,654,208
    const size_t szQK = (size_t)Bn * HEADS * SP * DP * 2;     // 36,175,872
    const size_t szV  = (size_t)Bn * HEADS * DVP * SP * 2;    // 30,146,560
    const size_t szC  = szX;

    char* ws = (char*)d_ws;
    unsigned short* Xbf = (unsigned short*)(ws);
    unsigned short* WqT = (unsigned short*)(ws + szX);
    unsigned short* WkT = (unsigned short*)(ws + szX + szW);
    unsigned short* WvT = (unsigned short*)(ws + szX + 2 * szW);
    unsigned short* WoT = (unsigned short*)(ws + szX + 3 * szW);
    unsigned short* Qh  = (unsigned short*)(ws + szX + 4 * szW);
    unsigned short* Kh  = (unsigned short*)((char*)Qh + szQK);
    unsigned short* Vt  = (unsigned short*)((char*)Kh + szQK);
    unsigned short* Ctx = (unsigned short*)((char*)Vt + szV);
    (void)szC; (void)ws_size; (void)in_sizes; (void)n_in; (void)out_size;

    // stage 0: pack + transpose weights + zero the padded Q/K/V region
    {
        int nX = MT * HID;
        pack_bf16<<<(nX + 255) / 256, 256, 0, stream>>>(X, Xbf, nX);
        int nW = HID * HID;
        int gW = (nW + 255) / 256;
        transpose_pack<<<gW, 256, 0, stream>>>(Wq, WqT);
        transpose_pack<<<gW, 256, 0, stream>>>(Wk, WkT);
        transpose_pack<<<gW, 256, 0, stream>>>(Wv, WvT);
        transpose_pack<<<gW, 256, 0, stream>>>(Wo, WoT);
        size_t n4 = (2 * szQK + szV) / 16;
        zero_u4<<<4096, 256, 0, stream>>>((uint4*)Qh, n4);
    }

    // stage 1-3: QKV projections (head-split / padded / V transposed)
    dim3 ggrid((MT + 127) / 128, HID / 64);
    gemm_bf16<<<ggrid, 256, 0, stream>>>(Xbf, WqT, bq, Qh, 0);
    gemm_bf16<<<ggrid, 256, 0, stream>>>(Xbf, WkT, bk, Kh, 0);
    gemm_bf16<<<ggrid, 256, 0, stream>>>(Xbf, WvT, bv, Vt, 1);

    // stage 4: attention (scores -> softmax -> PV)
    dim3 agrid(SP / 32, Bn * HEADS);
    size_t lds = (size_t)32 * SP * 4 + (size_t)32 * SP * 2;   // 141,312 B
    attn_kernel<<<agrid, 128, lds, stream>>>(Qh, Kh, Vt, Ctx);

    // stage 5: output projection -> f32 out
    gemm_bf16<<<ggrid, 256, 0, stream>>>(Ctx, WoT, bo, d_out, 2);
}
